// GIN_Node_35158602285141
// MI455X (gfx1250) — matrix operand order repaired and checked by reading, not compiled, not run on hardware
//
#include <hip/hip_runtime.h>

typedef __attribute__((ext_vector_type(16))) __bf16 v16bf;
typedef __attribute__((ext_vector_type(8)))  float  v8f;

#define D 128
#define SROWS 256

// ---------------------------------------------------------------- elementwise
__global__ void copy_f4(const float4* __restrict__ src, float4* __restrict__ dst, int n4) {
    int i = blockIdx.x * blockDim.x + threadIdx.x;
    if (i < n4) dst[i] = src[i];
}

// agg[dst[e]] += x[src[e]]  (one wave per edge, 4 features per lane)
__global__ void scatter_edges(const float* __restrict__ x,
                              const long long* __restrict__ ei,
                              float* __restrict__ agg, int E) {
    int gid  = blockIdx.x * blockDim.x + threadIdx.x;
    int edge = gid >> 5;
    if (edge >= E) return;
    int lane = gid & 31;
    long long s = ei[edge];
    long long d = ei[(long long)E + edge];
    float4 v = *(const float4*)(x + s * D + lane * 4);
    float* o = agg + d * D + lane * 4;
    atomicAdd(o + 0, v.x);
    atomicAdd(o + 1, v.y);
    atomicAdd(o + 2, v.z);
    atomicAdd(o + 3, v.w);
}

// ---------------------------------------------------------------- BN helpers
__global__ void zero256(float* p) { p[threadIdx.x] = 0.0f; }   // zeros ssum+ssq

__global__ void col_stats(const float* __restrict__ h, int M,
                          float* __restrict__ ssum, float* __restrict__ ssq) {
    int col = threadIdx.x;                 // 128 threads, coalesced row reads
    int r0 = blockIdx.x * SROWS;
    int r1 = r0 + SROWS; if (r1 > M) r1 = M;
    float s = 0.f, q = 0.f;
    for (int r = r0; r < r1; ++r) {
        float v = h[(long long)r * D + col];
        s += v; q += v * v;
    }
    atomicAdd(&ssum[col], s);
    atomicAdd(&ssq[col], q);
}

__global__ void bn_finalize(const float* __restrict__ ssum, const float* __restrict__ ssq,
                            const float* __restrict__ g, const float* __restrict__ be,
                            float invM, float* __restrict__ scale, float* __restrict__ shift) {
    int c = threadIdx.x;                   // 128 threads
    float mu  = ssum[c] * invM;
    float var = ssq[c] * invM - mu * mu;
    float rs  = rsqrtf(var + 1e-5f);
    float sc  = g[c] * rs;
    scale[c] = sc;
    shift[c] = be[c] - mu * sc;
}

// x = relu(bn(h)); agg = x  (agg may alias h: pure elementwise read-then-write)
__global__ void bn_relu_dual(const float* __restrict__ h,
                             const float* __restrict__ scale, const float* __restrict__ shift,
                             float* __restrict__ x, float* __restrict__ agg, long long total) {
    long long i = ((long long)blockIdx.x * blockDim.x + threadIdx.x) * 4;
    if (i >= total) return;
    int col = (int)(i & (D - 1));
    float4 hv = *(const float4*)(h + i);
    float4 r;
    r.x = fmaxf(hv.x * scale[col + 0] + shift[col + 0], 0.f);
    r.y = fmaxf(hv.y * scale[col + 1] + shift[col + 1], 0.f);
    r.z = fmaxf(hv.z * scale[col + 2] + shift[col + 2], 0.f);
    r.w = fmaxf(hv.w * scale[col + 3] + shift[col + 3], 0.f);
    *(float4*)(x + i)   = r;
    *(float4*)(agg + i) = r;
}

// ---------------------------------------------------------------- WMMA GEMMs
// Out[M,128] = relu( (A*scale+shift) @ W[128,128] + bias ), bf16 WMMA, f32 accum.
__global__ __launch_bounds__(256) void gemm_bn_relu_wmma(
    const float* __restrict__ A, const float* __restrict__ W,
    const float* __restrict__ bias,
    const float* __restrict__ scale, const float* __restrict__ shift,
    float* __restrict__ Out, int M)
{
    __shared__ __align__(32) __bf16 Wt[D * D];   // transposed: Wt[n*128 + k]
    __shared__ float sS[D], sT[D], sB[D];
    int tid = threadIdx.x;
    if (tid < D) {
        sS[tid] = scale ? scale[tid] : 1.0f;
        sT[tid] = shift ? shift[tid] : 0.0f;
        sB[tid] = bias[tid];
    }
    for (int i = tid; i < D * D; i += 256) {
        int k = i >> 7, n = i & (D - 1);
        Wt[n * D + k] = (__bf16)W[i];
    }
    __syncthreads();

    int wave = tid >> 5, lane = tid & 31;
    int rowBase = blockIdx.x * 128 + wave * 16;
    int mrow = rowBase + (lane & 15);
    int arow = mrow < M ? mrow : (M - 1);
    int koff = (lane >> 4) << 4;           // A frag: lanes<16 -> K 0..15
    const float* Arow = A + (long long)arow * D;

    v8f acc[8] = {};
    #pragma unroll
    for (int kb = 0; kb < D; kb += 32) {
        v16bf a;
        #pragma unroll
        for (int j = 0; j < 8; ++j) {
            int k0 = kb + koff + j;
            int k1 = k0 + 8;
            a[j]     = (__bf16)(Arow[k0] * sS[k0] + sT[k0]);
            a[8 + j] = (__bf16)(Arow[k1] * sS[k1] + sT[k1]);
        }
        #pragma unroll
        for (int nt = 0; nt < 8; ++nt) {
            int n = nt * 16 + (lane & 15);
            const v16bf* bp = (const v16bf*)&Wt[n * D + kb + ((lane >> 4) << 4)];
            acc[nt] = __builtin_amdgcn_wmma_f32_16x16x32_bf16(
                false, a, false, *bp, (short)0, acc[nt], false, false);
        }
    }

    #pragma unroll
    for (int nt = 0; nt < 8; ++nt) {
        int col = nt * 16 + (lane & 15);
        float b = sB[col];
        #pragma unroll
        for (int r = 0; r < 8; ++r) {
            int row = rowBase + r + ((lane >> 4) << 3);
            if (row < M) {
                float v = acc[nt][r] + b;
                Out[(long long)row * D + col] = v > 0.0f ? v : 0.0f;
            }
        }
    }
}

// Out[M,16] = A @ Wc[128,16] + bc
__global__ __launch_bounds__(256) void gemm_cls_wmma(
    const float* __restrict__ A, const float* __restrict__ Wc,
    const float* __restrict__ bc, float* __restrict__ Out, int M)
{
    __shared__ __align__(32) __bf16 Wt[16 * D];  // Wt[n*128 + k]
    __shared__ float sB[16];
    int tid = threadIdx.x;
    if (tid < 16) sB[tid] = bc[tid];
    for (int i = tid; i < 16 * D; i += 256) {
        int k = i >> 4, n = i & 15;
        Wt[n * D + k] = (__bf16)Wc[i];
    }
    __syncthreads();

    int wave = tid >> 5, lane = tid & 31;
    int rowBase = blockIdx.x * 128 + wave * 16;
    int mrow = rowBase + (lane & 15);
    int arow = mrow < M ? mrow : (M - 1);
    int koff = (lane >> 4) << 4;
    const float* Arow = A + (long long)arow * D;

    v8f acc = {};
    #pragma unroll
    for (int kb = 0; kb < D; kb += 32) {
        v16bf a;
        #pragma unroll
        for (int j = 0; j < 8; ++j) {
            a[j]     = (__bf16)Arow[kb + koff + j];
            a[8 + j] = (__bf16)Arow[kb + koff + 8 + j];
        }
        const v16bf* bp = (const v16bf*)&Wt[(lane & 15) * D + kb + ((lane >> 4) << 4)];
        acc = __builtin_amdgcn_wmma_f32_16x16x32_bf16(
            false, a, false, *bp, (short)0, acc, false, false);
    }

    int col = lane & 15;
    float b = sB[col];
    #pragma unroll
    for (int r = 0; r < 8; ++r) {
        int row = rowBase + r + ((lane >> 4) << 3);
        if (row < M) Out[(long long)row * 16 + col] = acc[r] + b;
    }
}

// ---------------------------------------------------------------- driver
extern "C" void kernel_launch(void* const* d_in, const int* in_sizes, int n_in,
                              void* d_out, int out_size, void* d_ws, size_t ws_size,
                              hipStream_t stream) {
    const float*     x_in = (const float*)d_in[0];
    const long long* ei   = (const long long*)d_in[2];   // int64 (2,E)
    const float* W1  = (const float*)d_in[3];
    const float* b1  = (const float*)d_in[4];
    const float* g1  = (const float*)d_in[5];
    const float* be1 = (const float*)d_in[6];
    const float* W2  = (const float*)d_in[7];
    const float* b2  = (const float*)d_in[8];
    const float* g2  = (const float*)d_in[9];
    const float* be2 = (const float*)d_in[10];
    const float* Wc  = (const float*)d_in[11];
    const float* bc  = (const float*)d_in[12];

    int N = in_sizes[0] / D;     // 50000
    int E = in_sizes[1];         // edge_attr is (E,1)

    float* B0   = (float*)d_ws;                 // agg / h2
    float* B1   = B0 + (size_t)N * D;           // h1 / x
    float* ssum = B1 + (size_t)N * D;           // [128]
    float* ssq  = ssum + D;                     // [128]
    float* scl  = ssq + D;                      // [128]
    float* shf  = scl + D;                      // [128]
    float invM = 1.0f / (float)N;

    long long total = (long long)N * D;
    int gGemm    = (N + 127) / 128;
    int gStats   = (N + SROWS - 1) / SROWS;
    int gElem    = (int)((total / 4 + 255) / 256);
    int gScatter = (int)(((long long)E * 32 + 255) / 256);

    for (int l = 0; l < 3; ++l) {
        const float* xc = (l == 0) ? x_in : B1;
        if (l == 0)  // agg = x (later layers: bn_relu_dual already wrote agg=x)
            copy_f4<<<gElem, 256, 0, stream>>>((const float4*)x_in, (float4*)B0,
                                               (int)(total / 4));
        scatter_edges<<<gScatter, 256, 0, stream>>>(xc, ei, B0, E);

        // h1 = relu(agg @ W1 + b1)
        gemm_bn_relu_wmma<<<gGemm, 256, 0, stream>>>(
            B0, W1 + (size_t)l * D * D, b1 + l * D, nullptr, nullptr, B1, N);
        zero256<<<1, 256, 0, stream>>>(ssum);               // zeros ssum+ssq
        col_stats<<<gStats, D, 0, stream>>>(B1, N, ssum, ssq);
        bn_finalize<<<1, D, 0, stream>>>(ssum, ssq, g1 + l * D, be1 + l * D,
                                         invM, scl, shf);

        // h2 = relu(bn(h1) @ W2 + b2)   (BN fused into A-load)
        gemm_bn_relu_wmma<<<gGemm, 256, 0, stream>>>(
            B1, W2 + (size_t)l * D * D, b2 + l * D, scl, shf, B0, N);
        zero256<<<1, 256, 0, stream>>>(ssum);
        col_stats<<<gStats, D, 0, stream>>>(B0, N, ssum, ssq);
        bn_finalize<<<1, D, 0, stream>>>(ssum, ssq, g2 + l * D, be2 + l * D,
                                         invM, scl, shf);

        // x = relu(bn(h2));  agg = x  (in-place on B0 for agg, B1 for x)
        bn_relu_dual<<<gElem, 256, 0, stream>>>(B0, scl, shf, B1, B0, total);
    }

    // out = x @ Wc + bc
    gemm_cls_wmma<<<gGemm, 256, 0, stream>>>(B1, Wc, bc, (float*)d_out, N);
}
